// EnTransformerBlock_79413945303877
// MI455X (gfx1250) — compile-verified, matrix operand order checked by reading, not run on hardware
//
// EnTransformerBlock for MI455X (gfx1250): fully fused per-receiver flash-attention
// style pipeline. WMMA f16->f32 for all GEMMs, TDM async tensor loads for the
// sender-position tiles, async global->LDS for the Q row, online softmax in LDS.
#include <hip/hip_runtime.h>
#include <hip/hip_bf16.h>

typedef _Float16 half_t;
typedef __attribute__((ext_vector_type(16))) _Float16 v16h;
typedef __attribute__((ext_vector_type(8)))  _Float16 v8h;
typedef __attribute__((ext_vector_type(8)))  float    v8f;
typedef __attribute__((ext_vector_type(4)))  unsigned int v4u;
typedef __attribute__((ext_vector_type(4)))  int      v4i;
typedef __attribute__((ext_vector_type(8)))  int      v8i;

// Exact parameter types for __builtin_amdgcn_global_load_async_to_lds_b128:
// GCC-style int vector of 4, in addrspace(1) (global) and addrspace(3) (LDS).
typedef int v4i_vs __attribute__((vector_size(16)));
typedef __attribute__((address_space(1))) v4i_vs as1_v4i;
typedef __attribute__((address_space(3))) v4i_vs as3_v4i;

#if __has_builtin(__builtin_amdgcn_tensor_load_to_lds) && __has_builtin(__builtin_amdgcn_s_wait_tensorcnt)
#define USE_TDM 1
#else
#define USE_TDM 0
#endif

#if __has_builtin(__builtin_amdgcn_global_load_async_to_lds_b128) && __has_builtin(__builtin_amdgcn_s_wait_asynccnt)
#define USE_ASYNC 1
#else
#define USE_ASYNC 0
#endif

// Compile-loop diagnostics: report which CDNA5 data-mover paths are active in
// the DEVICE pass (shows up in hipcc stderr even on success).
#ifdef __HIP_DEVICE_COMPILE__
#define STR2(x) #x
#define STR(x) STR2(x)
#pragma message("CDNA5 device pass, clang major " STR(__clang_major__))
#if USE_TDM
#pragma message("CDNA5: TDM tensor_load_to_lds path ENABLED")
#else
#pragma message("CDNA5: TDM tensor_load_to_lds path DISABLED")
#endif
#if USE_ASYNC
#pragma message("CDNA5: global_load_async_to_lds path ENABLED")
#else
#pragma message("CDNA5: global_load_async_to_lds path DISABLED")
#endif
#endif

// ---------------------------------------------------------------------------
// WMMA helpers (v_wmma_f32_16x16x32_f16). A: 16x32 f16 per wave, B: 32x16 f16,
// C/D: 16x16 f32 (8 VGPRs). Layouts per CDNA5 ISA 7.12.2.
// ---------------------------------------------------------------------------
__device__ __forceinline__ v8f wmma_f16(v16h a, v16h b, v8f c) {
  return __builtin_amdgcn_wmma_f32_16x16x32_f16(false, a, false, b, (short)0, c,
                                                false, false);
}

__device__ __forceinline__ v8f zero8() {
  v8f c;
#pragma unroll
  for (int i = 0; i < 8; ++i) c[i] = 0.0f;
  return c;
}

// A tile (16 rows x 32 K) from an LDS row-major f16 matrix.
// lane: m = rowBase + (lane&15); K slots: k = 32*ks + (e&7) + 16*(e>=8) + 8*(lane>=16)
// -> two contiguous 16B LDS loads per tile.
__device__ __forceinline__ v16h load_a_lds(const half_t* sh, int rowBase,
                                           int strideH, int ks) {
  const int lane = threadIdx.x & 31;
  const int m = rowBase + (lane & 15);
  const int k0 = (ks << 5) + ((lane >> 4) << 3);
  const half_t* p = sh + (size_t)m * strideH + k0;
  v8h lo = *(const v8h*)p;
  v8h hi = *(const v8h*)(p + 16);
  v16h a;
#pragma unroll
  for (int i = 0; i < 8; ++i) { a[i] = lo[i]; a[i + 8] = hi[i]; }
  return a;
}

// B tile from a pre-packed weight image (see pack_b_kernel): per (coltile c, kstep)
// the 32 lanes' 16 halves are stored lane-major -> contiguous 32B per lane.
__device__ __forceinline__ v16h load_b_packed(const half_t* pk, int c, int KS,
                                              int ks) {
  const int lane = threadIdx.x & 31;
  const half_t* p = pk + ((size_t)((c * KS + ks) << 5) + lane) * 16;
  v8h lo = *(const v8h*)p;
  v8h hi = *(const v8h*)(p + 8);
  v16h b;
#pragma unroll
  for (int i = 0; i < 8; ++i) { b[i] = lo[i]; b[i + 8] = hi[i]; }
  return b;
}

// C/D layout: VGPR g holds row m = rowBase + g + 8*(lane>=16), col n = 16c+(lane&15)
__device__ __forceinline__ void store_c_silu(v8f C, half_t* out, int rowBase,
                                             int strideO, int c,
                                             const float* bias) {
  const int lane = threadIdx.x & 31;
  const int n = (c << 4) + (lane & 15);
  const int m0 = rowBase + ((lane >> 4) << 3);
  const float b = bias[n];
#pragma unroll
  for (int g = 0; g < 8; ++g) {
    float x = C[g] + b;
    out[(size_t)(m0 + g) * strideO + n] = (half_t)(x / (1.0f + __expf(-x)));
  }
}

// Full 128-row x 128-col GEMM with silu epilogue; wave == column tile.
template <int KS>
__device__ __forceinline__ void gemm128_silu(const half_t* shA, int strideA,
                                             const half_t* Bpk, const float* bias,
                                             half_t* shOut, int strideO) {
  const int c = threadIdx.x >> 5;
  v16h B[KS];
#pragma unroll
  for (int ks = 0; ks < KS; ++ks) B[ks] = load_b_packed(Bpk, c, KS, ks);
#pragma unroll
  for (int rt = 0; rt < 8; ++rt) {
    v8f C = zero8();
#pragma unroll
    for (int ks = 0; ks < KS; ++ks)
      C = wmma_f16(load_a_lds(shA, rt << 4, strideA, ks), B[ks], C);
    store_c_silu(C, shOut, rt << 4, strideO, c, bias);
  }
}

// 16-row tile GEMM returning the accumulator (wave == column tile).
template <int KS>
__device__ __forceinline__ v8f gemm_tile16(const half_t* shA, int strideA,
                                           const half_t* Bpk) {
  const int c = threadIdx.x >> 5;
  v16h B[KS];
#pragma unroll
  for (int ks = 0; ks < KS; ++ks) B[ks] = load_b_packed(Bpk, c, KS, ks);
  v8f C = zero8();
#pragma unroll
  for (int ks = 0; ks < KS; ++ks)
    C = wmma_f16(load_a_lds(shA, 0, strideA, ks), B[ks], C);
  return C;
}

// ---------------------------------------------------------------------------
// TDM: 2D tile (rows x elems_per_row of 4-byte elements) Global -> LDS.
// D# per CDNA5 ISA ch.8 (group0: flags/lds/global addr, group1: dims/strides).
// ---------------------------------------------------------------------------
#if USE_TDM
__device__ __forceinline__ void tdm_load_2d(unsigned lds_off, const void* gptr,
                                            unsigned elems_per_row, unsigned rows,
                                            unsigned row_stride_elems) {
  unsigned long long ga = (unsigned long long)(uintptr_t)gptr;
  v4u g0;
  g0[0] = 1u;                                            // count=1, user D#
  g0[1] = lds_off;                                       // lds_addr
  g0[2] = (unsigned)(ga & 0xFFFFFFFFu);                  // global_addr[31:0]
  g0[3] = (unsigned)((ga >> 32) & 0x01FFFFFFu) | (2u << 30);  // addr[56:32], type=2
  v8i g1;
  const unsigned dsz = 2u;                               // 4-byte elements
  g1[0] = (int)(dsz << 16);                              // wg_mask=0, data_size
  g1[1] = (int)((elems_per_row & 0xFFFFu) << 16);        // tensor_dim0[15:0]
  g1[2] = (int)((elems_per_row >> 16) | ((rows & 0xFFFFu) << 16));  // dim0 hi | dim1 lo
  g1[3] = (int)((rows >> 16) | ((elems_per_row & 0xFFFFu) << 16));  // dim1 hi | tile_dim0
  g1[4] = (int)(rows & 0xFFFFu);                         // tile_dim1, tile_dim2=0
  g1[5] = (int)row_stride_elems;                         // tensor_dim0_stride[31:0]
  g1[6] = 0;
  g1[7] = 0;
  v4i z4; z4[0] = z4[1] = z4[2] = z4[3] = 0;
#if __clang_major__ >= 23
  v8i z8;
#pragma unroll
  for (int i = 0; i < 8; ++i) z8[i] = 0;
  __builtin_amdgcn_tensor_load_to_lds(g0, g1, z4, z4, z8, 0);
#else
  __builtin_amdgcn_tensor_load_to_lds(g0, g1, z4, z4, 0);
#endif
}
#endif

#if USE_ASYNC
// One 16-byte async Global->LDS copy per lane (ASYNCcnt tracked).
// Builtin signature: (v4i addrspace(1)*, v4i addrspace(3)*, imm offset, imm cpol)
__device__ __forceinline__ void async_g2l_b128(const void* g, void* l) {
  __builtin_amdgcn_global_load_async_to_lds_b128(
      (as1_v4i*)(uintptr_t)g, (as3_v4i*)(uintptr_t)l, 0, 0);
}
#endif

// ---------------------------------------------------------------------------
// Prep kernels: weight packing into the per-lane WMMA B layout, f32->f16 copy.
// packed index i = (((c*KS + ks)*32) + lane)*16 + e  maps to  W[k, n] with
// n = 16c + (lane&15),  k = 32ks + (e&7) + 16*(e>=8) + 8*(lane>=16)
// ---------------------------------------------------------------------------
__global__ void pack_b_kernel(const float* __restrict__ W, half_t* __restrict__ dst,
                              int K, int Nreal, int Npad) {
  const int i = blockIdx.x * blockDim.x + threadIdx.x;
  const int total = K * Npad;
  if (i >= total) return;
  const int e = i & 15;
  const int lane = (i >> 4) & 31;
  const int t = i >> 9;
  const int KS = K >> 5;
  const int ks = t % KS;
  const int c = t / KS;
  const int n = (c << 4) + (lane & 15);
  const int k = (ks << 5) + (e & 7) + ((e >> 3) << 4) + ((lane >> 4) << 3);
  const float v = (n < Nreal) ? W[(size_t)k * Nreal + n] : 0.0f;
  dst[i] = (half_t)v;
}

__global__ void f2h_kernel(const float* __restrict__ src, half_t* __restrict__ dst,
                           int n) {
  const int i = blockIdx.x * blockDim.x + threadIdx.x;
  if (i < n) dst[i] = (half_t)src[i];
}

// ---------------------------------------------------------------------------
// Kernel 2: node projections Qn = feat@Wq, Kn = feat@Wk, Vn = feat@Wv (f32 out)
// ---------------------------------------------------------------------------
__global__ void __launch_bounds__(256)
node_proj_kernel(const half_t* __restrict__ featF16,
                 const half_t* __restrict__ Wqp, const half_t* __restrict__ Wkp,
                 const half_t* __restrict__ Wvp, float* __restrict__ Qn,
                 float* __restrict__ Kn, float* __restrict__ Vn) {
  __shared__ __attribute__((aligned(16))) half_t shA[16 * 128];
  const int tid = threadIdx.x;
  const int node0 = blockIdx.x * 16;
  *(v8h*)(shA + tid * 8) = *(const v8h*)(featF16 + (size_t)node0 * 128 + tid * 8);
  __syncthreads();
  const int lane = tid & 31;
  const int c = tid >> 5;
  const int n = lane & 15;
  const int m0 = (lane >> 4) << 3;
  const half_t* Bp[3] = {Wqp, Wkp, Wvp};
  float* Op[3] = {Qn, Kn, Vn};
#pragma unroll
  for (int mat = 0; mat < 3; ++mat) {
    v8f C = gemm_tile16<4>(shA, 128, Bp[mat]);
#pragma unroll
    for (int g = 0; g < 8; ++g)
      Op[mat][(size_t)(node0 + m0 + g) * 128 + (c << 4) + n] = C[g];
  }
}

// ---------------------------------------------------------------------------
// Kernel 3: fused per-receiver edge pipeline + online softmax aggregation.
// One block (8 wave32) per receiver; 4 sender tiles of 128.
// ---------------------------------------------------------------------------
__global__ void __launch_bounds__(256)
edge_attn_kernel(const float* __restrict__ positions,
                 const float* __restrict__ Qn, const float* __restrict__ Kn,
                 const float* __restrict__ Vn,
                 const half_t* __restrict__ We0p, const float* __restrict__ be0,
                 const half_t* __restrict__ We1p, const float* __restrict__ be1,
                 const half_t* __restrict__ Wekp, const half_t* __restrict__ Wevp,
                 const half_t* __restrict__ Wgp, float* __restrict__ aggS,
                 float* __restrict__ outPos) {
  __shared__ __attribute__((aligned(16))) half_t shEscal[128 * 64];  // 16 KB
  __shared__ __attribute__((aligned(16))) half_t shH1[128 * 128];    // 32 KB
  __shared__ __attribute__((aligned(16))) half_t shH2[128 * 128];    // 32 KB
  __shared__ __attribute__((aligned(16))) float shPos[128 * 24];     // 12 KB
  __shared__ __attribute__((aligned(16))) float shQ[128];
  __shared__ float shUnit[128 * 8 * 3];                              // 12 KB
  __shared__ float shGate[128 * 8];
  __shared__ float shLG[128 * 8];
  __shared__ float shAccS[128];
  __shared__ float shAccV[24];
  __shared__ float shM[8];
  __shared__ float shD[8];

  const int tid = threadIdx.x;
  const int lane = tid & 31;
  const int wave = tid >> 5;
  const int r = blockIdx.x;

  // Receiver's Q row: async global->LDS DMA (wave 0, 32 lanes x 16B = 512B)
#if USE_ASYNC
  if (tid < 32) {
    async_g2l_b128(Qn + (size_t)r * 128 + lane * 4, shQ + lane * 4);
    __builtin_amdgcn_s_wait_asynccnt(0);
  }
  if (tid < 128) shAccS[tid] = 0.0f;
#else
  if (tid < 128) { shQ[tid] = Qn[(size_t)r * 128 + tid]; shAccS[tid] = 0.0f; }
#endif
  if (tid < 24) shAccV[tid] = 0.0f;
  if (tid < 8) { shM[tid] = -1e30f; shD[tid] = 0.0f; }
  __syncthreads();

  for (int st = 0; st < 4; ++st) {
    const int sBase = st * 128;

    // --- sender position tile -> LDS via Tensor Data Mover ---
#if USE_TDM
    if (tid < 32) {  // wave 0 issues the DMA; TDM ignores EXEC
      tdm_load_2d((unsigned)(uintptr_t)&shPos[0],
                  positions + (size_t)sBase * 24, 24u, 128u, 24u);
      __builtin_amdgcn_s_wait_tensorcnt(0);
    }
#else
    for (int i = tid; i < 128 * 24; i += 256)
      shPos[i] = positions[(size_t)sBase * 24 + i];
#endif
    if (tid == 0) {  // warm L2/GL1 for the K/V node tiles used below
      __builtin_prefetch(Kn + (size_t)sBase * 128, 0, 1);
      __builtin_prefetch(Vn + (size_t)sBase * 128, 0, 1);
    }
    __syncthreads();

    // --- geometry + bessel*envelope radial features (only l=1 sph is used) ---
#pragma unroll
    for (int it = 0; it < 4; ++it) {
      const int pair = tid + it * 256;  // 1024 (sender, vchan) pairs
      const int j = pair >> 3;
      const int v = pair & 7;
      const float px = shPos[j * 24 + v * 3 + 0];
      const float py = shPos[j * 24 + v * 3 + 1];
      const float pz = shPos[j * 24 + v * 3 + 2];
      const float rx = positions[(size_t)r * 24 + v * 3 + 0];
      const float ry = positions[(size_t)r * 24 + v * 3 + 1];
      const float rz = positions[(size_t)r * 24 + v * 3 + 2];
      const float dx = px - rx, dy = py - ry, dz = pz - rz;
      const float len = sqrtf(dx * dx + dy * dy + dz * dz + 1e-20f);
      const float inv = 1.0f / len;
      const float s3 = 1.7320508075688772f;  // sph l=1 'component' norm
      shUnit[(j * 8 + v) * 3 + 0] = s3 * dx * inv;
      shUnit[(j * 8 + v) * 3 + 1] = s3 * dy * inv;
      shUnit[(j * 8 + v) * 3 + 2] = s3 * dz * inv;
      const float arg = 2.0f * (1.0f - 0.1f * len);
      const float env = (arg > 0.0f) ? 1.2f * __expf(0.5f - 1.0f / arg) : 0.0f;
      const float amp = 0.4472135955f * inv * env;  // sqrt(2/10)/x * envelope
      // sin(k*pi*len/10), k=1..8 via Chebyshev recurrence: 1 sincos + 2 FMA/term
      float s1, c1;
      __sincosf(0.31415926535897932f * len, &s1, &c1);
      const float twoC = 2.0f * c1;
      float skm1 = 0.0f, sk = s1;
#pragma unroll
      for (int b = 0; b < 8; ++b) {
        shEscal[j * 64 + v * 8 + b] = (half_t)(amp * sk);
        const float skp1 = twoC * sk - skm1;
        skm1 = sk;
        sk = skp1;
      }
    }
    __syncthreads();

    // --- edge MLP: h = silu(silu(escal@We0+be0)@We1+be1) ---
    gemm128_silu<2>(shEscal, 64, We0p, be0, shH1, 128);
    __syncthreads();
    gemm128_silu<4>(shH1, 128, We1p, be1, shH2, 128);
    __syncthreads();

    // --- gates: gate = h @ Wg (N padded 8->16); wave == row tile ---
    {
      const int rt = wave;
      v16h Bg[4];
#pragma unroll
      for (int ks = 0; ks < 4; ++ks) Bg[ks] = load_b_packed(Wgp, 0, 4, ks);
      v8f C = zero8();
#pragma unroll
      for (int ks = 0; ks < 4; ++ks)
        C = wmma_f16(load_a_lds(shH2, rt << 4, 128, ks), Bg[ks], C);
      const int n = lane & 15;
      const int m0 = (rt << 4) + ((lane >> 4) << 3);
      if (n < 8) {
#pragma unroll
        for (int g = 0; g < 8; ++g) shGate[(m0 + g) * 8 + n] = C[g];
      }
    }

    // --- keys: k_e = h@Wek + Kn[s]; wave == head; logits via lane reduce ---
    {
      const int c = wave;
      const int n = lane & 15;
      v16h Bk[4];
#pragma unroll
      for (int ks = 0; ks < 4; ++ks) Bk[ks] = load_b_packed(Wekp, c, 4, ks);
      const float qv = shQ[(c << 4) + n];
#pragma unroll
      for (int rt = 0; rt < 8; ++rt) {
        v8f C = zero8();
#pragma unroll
        for (int ks = 0; ks < 4; ++ks)
          C = wmma_f16(load_a_lds(shH2, rt << 4, 128, ks), Bk[ks], C);
        const int m0 = (rt << 4) + ((lane >> 4) << 3);
#pragma unroll
        for (int g = 0; g < 8; ++g) {
          const int j = m0 + g;
          const int s = sBase + j;
          float t = (C[g] + Kn[(size_t)s * 128 + (c << 4) + n]) * qv;
          t += __shfl_xor(t, 1, 32);   // reduce over 16 feature lanes
          t += __shfl_xor(t, 2, 32);
          t += __shfl_xor(t, 4, 32);
          t += __shfl_xor(t, 8, 32);
          if (n == 0) {
            float lg = t * 0.25f;                 // 1/sqrt(FH)
            if (s == r) lg = -1e30f;              // no self edge
            shLG[j * 8 + c] = lg;
          }
        }
      }
    }
    __syncthreads();

    // --- online softmax per head (wave == head) ---
    {
      const int c = wave;
      const float mOld = shM[c];
      float lmax = -1e30f;
      for (int j = lane; j < 128; j += 32) lmax = fmaxf(lmax, shLG[j * 8 + c]);
#pragma unroll
      for (int m = 1; m < 32; m <<= 1) lmax = fmaxf(lmax, __shfl_xor(lmax, m, 32));
      const float mNew = fmaxf(mOld, lmax);
      const float scale = __expf(mOld - mNew);
      float sum = 0.0f;
      for (int j = lane; j < 128; j += 32) {
        const float e = __expf(shLG[j * 8 + c] - mNew);
        shLG[j * 8 + c] = e;  // keep unnormalized weight
        sum += e;
      }
#pragma unroll
      for (int m = 1; m < 32; m <<= 1) sum += __shfl_xor(sum, m, 32);
      if (lane == 0) { shD[c] = shD[c] * scale + sum; shM[c] = mNew; }
      if (lane < 16) shAccS[(c << 4) + lane] *= scale;
      if (lane < 3) shAccV[c * 3 + lane] *= scale;
    }
    __syncthreads();

    // --- values: v_e = h@Wev + Vn[s]; weighted accumulate (wave == head) ---
    {
      const int c = wave;
      const int n = lane & 15;
      v16h Bv[4];
#pragma unroll
      for (int ks = 0; ks < 4; ++ks) Bv[ks] = load_b_packed(Wevp, c, 4, ks);
      float part = 0.0f;
#pragma unroll
      for (int rt = 0; rt < 8; ++rt) {
        v8f C = zero8();
#pragma unroll
        for (int ks = 0; ks < 4; ++ks)
          C = wmma_f16(load_a_lds(shH2, rt << 4, 128, ks), Bv[ks], C);
        const int m0 = (rt << 4) + ((lane >> 4) << 3);
#pragma unroll
        for (int g = 0; g < 8; ++g) {
          const int j = m0 + g;
          const float vv = C[g] + Vn[(size_t)(sBase + j) * 128 + (c << 4) + n];
          part += shLG[j * 8 + c] * vv;
        }
      }
      part += __shfl_xor(part, 16, 32);  // combine the two row-halves
      if (lane < 16) shAccS[(c << 4) + lane] += part;

      // equivariant accumulate: vector channel v == head c (VH==1)
      float p0 = 0.0f, p1 = 0.0f, p2 = 0.0f;
      for (int j = lane; j < 128; j += 32) {
        const float w = shLG[j * 8 + c] * shGate[j * 8 + c];
        p0 += w * shUnit[(j * 8 + c) * 3 + 0];
        p1 += w * shUnit[(j * 8 + c) * 3 + 1];
        p2 += w * shUnit[(j * 8 + c) * 3 + 2];
      }
#pragma unroll
      for (int m = 1; m < 32; m <<= 1) {
        p0 += __shfl_xor(p0, m, 32);
        p1 += __shfl_xor(p1, m, 32);
        p2 += __shfl_xor(p2, m, 32);
      }
      if (lane == 0) {
        shAccV[c * 3 + 0] += p0;
        shAccV[c * 3 + 1] += p1;
        shAccV[c * 3 + 2] += p2;
      }
    }
    __syncthreads();
  }

  // --- finalize: normalize by softmax denominator, write outputs ---
  if (tid < 128) aggS[(size_t)r * 128 + tid] = shAccS[tid] / shD[tid >> 4];
  if (tid < 24)
    outPos[(size_t)r * 24 + tid] =
        positions[(size_t)r * 24 + tid] + shAccV[tid] / shD[tid / 3];
}

// ---------------------------------------------------------------------------
// Kernel 4: output MLP on agg_s + residual: features + MLP(agg_s)
// ---------------------------------------------------------------------------
__global__ void __launch_bounds__(256)
out_mlp_kernel(const float* __restrict__ aggS, const float* __restrict__ features,
               const half_t* __restrict__ Wo0p, const float* __restrict__ bo0,
               const half_t* __restrict__ Wo1p, const float* __restrict__ bo1,
               const half_t* __restrict__ Wo2p, const float* __restrict__ bo2,
               float* __restrict__ outFeat) {
  __shared__ __attribute__((aligned(16))) half_t sh1[16 * 128];
  __shared__ __attribute__((aligned(16))) half_t sh2[16 * 128];
  const int tid = threadIdx.x;
  const int node0 = blockIdx.x * 16;
#pragma unroll
  for (int i = 0; i < 8; ++i)
    sh1[tid * 8 + i] = (half_t)aggS[(size_t)node0 * 128 + tid * 8 + i];
  __syncthreads();
  const int lane = tid & 31;
  const int c = tid >> 5;
  {
    v8f C = gemm_tile16<4>(sh1, 128, Wo0p);
    store_c_silu(C, sh2, 0, 128, c, bo0);
  }
  __syncthreads();
  {
    v8f C = gemm_tile16<4>(sh2, 128, Wo1p);
    store_c_silu(C, sh1, 0, 128, c, bo1);
  }
  __syncthreads();
  {
    v8f C = gemm_tile16<4>(sh1, 128, Wo2p);
    const int n = (c << 4) + (lane & 15);
    const int m0 = (lane >> 4) << 3;
    const float b = bo2[n];
#pragma unroll
    for (int g = 0; g < 8; ++g) {
      const int row = node0 + m0 + g;
      outFeat[(size_t)row * 128 + n] =
          C[g] + b + features[(size_t)row * 128 + n];
    }
  }
}

// ---------------------------------------------------------------------------
// Host launcher
// ---------------------------------------------------------------------------
extern "C" void kernel_launch(void* const* d_in, const int* in_sizes, int n_in,
                              void* d_out, int out_size, void* d_ws, size_t ws_size,
                              hipStream_t stream) {
  (void)in_sizes; (void)n_in; (void)out_size; (void)ws_size;
  const float* positions = (const float*)d_in[0];
  const float* features  = (const float*)d_in[1];
  const float* We0 = (const float*)d_in[2];  const float* be0 = (const float*)d_in[3];
  const float* We1 = (const float*)d_in[4];  const float* be1 = (const float*)d_in[5];
  const float* Wq  = (const float*)d_in[6];
  const float* Wk  = (const float*)d_in[7];
  const float* Wek = (const float*)d_in[8];
  const float* Wv  = (const float*)d_in[9];
  const float* Wev = (const float*)d_in[10];
  const float* Wg  = (const float*)d_in[11];
  const float* Wo0 = (const float*)d_in[12]; const float* bo0 = (const float*)d_in[13];
  const float* Wo1 = (const float*)d_in[14]; const float* bo1 = (const float*)d_in[15];
  const float* Wo2 = (const float*)d_in[16]; const float* bo2 = (const float*)d_in[17];

  char* ws = (char*)d_ws;
  size_t off = 0;
  auto take = [&](size_t bytes) -> char* {
    char* p = ws + off;
    off += (bytes + 255) & ~(size_t)255;
    return p;
  };

  half_t* featF16 = (half_t*)take(512 * 128 * sizeof(half_t));
  half_t* We0p = (half_t*)take(64 * 128 * sizeof(half_t));
  half_t* We1p = (half_t*)take(128 * 128 * sizeof(half_t));
  half_t* Wqp  = (half_t*)take(128 * 128 * sizeof(half_t));
  half_t* Wkp  = (half_t*)take(128 * 128 * sizeof(half_t));
  half_t* Wvp  = (half_t*)take(128 * 128 * sizeof(half_t));
  half_t* Wekp = (half_t*)take(128 * 128 * sizeof(half_t));
  half_t* Wevp = (half_t*)take(128 * 128 * sizeof(half_t));
  half_t* Wgp  = (half_t*)take(128 * 16 * sizeof(half_t));
  half_t* Wo0p = (half_t*)take(128 * 128 * sizeof(half_t));
  half_t* Wo1p = (half_t*)take(128 * 128 * sizeof(half_t));
  half_t* Wo2p = (half_t*)take(128 * 128 * sizeof(half_t));
  float* Qn   = (float*)take(512 * 128 * sizeof(float));
  float* Kn   = (float*)take(512 * 128 * sizeof(float));
  float* Vn   = (float*)take(512 * 128 * sizeof(float));
  float* aggS = (float*)take(512 * 128 * sizeof(float));

  f2h_kernel<<<(512 * 128 + 255) / 256, 256, 0, stream>>>(features, featF16,
                                                          512 * 128);
  pack_b_kernel<<<(64 * 128 + 255) / 256, 256, 0, stream>>>(We0, We0p, 64, 128, 128);
  pack_b_kernel<<<(128 * 128 + 255) / 256, 256, 0, stream>>>(We1, We1p, 128, 128, 128);
  pack_b_kernel<<<(128 * 128 + 255) / 256, 256, 0, stream>>>(Wq, Wqp, 128, 128, 128);
  pack_b_kernel<<<(128 * 128 + 255) / 256, 256, 0, stream>>>(Wk, Wkp, 128, 128, 128);
  pack_b_kernel<<<(128 * 128 + 255) / 256, 256, 0, stream>>>(Wv, Wvp, 128, 128, 128);
  pack_b_kernel<<<(128 * 128 + 255) / 256, 256, 0, stream>>>(Wek, Wekp, 128, 128, 128);
  pack_b_kernel<<<(128 * 128 + 255) / 256, 256, 0, stream>>>(Wev, Wevp, 128, 128, 128);
  pack_b_kernel<<<(128 * 16 + 255) / 256, 256, 0, stream>>>(Wg, Wgp, 128, 8, 16);
  pack_b_kernel<<<(128 * 128 + 255) / 256, 256, 0, stream>>>(Wo0, Wo0p, 128, 128, 128);
  pack_b_kernel<<<(128 * 128 + 255) / 256, 256, 0, stream>>>(Wo1, Wo1p, 128, 128, 128);
  pack_b_kernel<<<(128 * 128 + 255) / 256, 256, 0, stream>>>(Wo2, Wo2p, 128, 128, 128);

  node_proj_kernel<<<32, 256, 0, stream>>>(featF16, Wqp, Wkp, Wvp, Qn, Kn, Vn);

  float* outPos = (float*)d_out;              // [512, 8, 3]
  float* outFeat = (float*)d_out + 512 * 24;  // [512, 128]
  edge_attn_kernel<<<512, 256, 0, stream>>>(positions, Qn, Kn, Vn, We0p, be0,
                                            We1p, be1, Wekp, Wevp, Wgp, aggS,
                                            outPos);
  out_mlp_kernel<<<32, 256, 0, stream>>>(aggS, features, Wo0p, bo0, Wo1p, bo1,
                                         Wo2p, bo2, outFeat);
}
